// GraphNeuralNetwork_58213986729999
// MI455X (gfx1250) — matrix operand order repaired and checked by reading, not compiled, not run on hardware
//
#include <hip/hip_runtime.h>

#define NN 100000   // nodes
#define NE 1600000  // edges
#define KD 128      // input/hidden dim (K of every GEMM)

typedef __bf16 v16bf __attribute__((ext_vector_type(16)));
typedef float  v8f   __attribute__((ext_vector_type(8)));

__device__ __forceinline__ unsigned short f2bf(float f) {
  union { float f; unsigned int u; } x; x.f = f;
  unsigned int r = x.u + 0x7FFFu + ((x.u >> 16) & 1u);   // round-to-nearest-even
  return (unsigned short)(r >> 16);
}
__device__ __forceinline__ float bf2f(unsigned short b) {
  union { float f; unsigned int u; } x; x.u = ((unsigned int)b) << 16;
  return x.f;
}

// ---------------- degree / dinv ----------------
__global__ void __launch_bounds__(256) k_deg_init(float* deg) {
  int n = blockIdx.x * 256 + threadIdx.x;
  if (n < NN) deg[n] = 1.0f;                 // self-loop contribution
}
__global__ void __launch_bounds__(256) k_deg_count(const int* __restrict__ col, float* deg) {
  int e = blockIdx.x * 256 + threadIdx.x;
  if (e < NE) unsafeAtomicAdd(&deg[col[e]], 1.0f);
}
__global__ void __launch_bounds__(256) k_dinv(float* deg) {
  int n = blockIdx.x * 256 + threadIdx.x;
  if (n < NN) deg[n] = rsqrtf(deg[n]);       // deg >= 1 always
}

// ---------------- conversions ----------------
// layer-1 input: xh[n,k] = bf16(x[n,k] * dinv[n])   (pre-scaled by source dinv)
__global__ void __launch_bounds__(256) k_cvt_x(const float* __restrict__ x,
                                               const float* __restrict__ dinv,
                                               unsigned short* __restrict__ xh) {
  int t = blockIdx.x * 256 + threadIdx.x;    // NN*KD total, exact multiple of 256
  xh[t] = f2bf(x[t] * dinv[t >> 7]);
}
// W [K=128, Nout] f32 row-major  ->  wcm [Nout, K] bf16 (K contiguous per output col)
__global__ void __launch_bounds__(256) k_cvt_w(const float* __restrict__ W,
                                               unsigned short* __restrict__ wcm,
                                               int Nout) {
  int t = blockIdx.x * 256 + threadIdx.x;    // Nout*KD total
  if (t >= Nout * KD) return;
  int n = t >> 7, k = t & 127;
  wcm[t] = f2bf(W[k * Nout + n]);
}

// ---------------- WMMA GEMM:  y = Xh(bf16) @ Wcm(bf16), f32 accum ----------------
// one wave per 16-row stripe: A fragments (16x128) loaded once into 32 VGPRs,
// then reused for every 16-col tile (Nout/16 of them). K=128 -> 4 WMMA per tile.
// epilogue writes bf16 messages (ybf) and f32 agg initialized with the
// self-loop term (agg = y; finalize later applies *dinv[c] + bias).
union BFrag { v16bf v; unsigned int u[8]; };

__global__ void __launch_bounds__(32) k_gemm_wmma(const unsigned short* __restrict__ xh,
                                                  const unsigned short* __restrict__ wcm,
                                                  unsigned short* __restrict__ ybf,
                                                  float* __restrict__ agg,
                                                  int Nout) {
  const int lane = threadIdx.x & 31;
  const int lm   = lane & 15;      // row (A) / col (B,C) within tile
  const int hh   = lane >> 4;      // lane-half selects K/M sub-block
  const int row0 = blockIdx.x << 4;

  const unsigned int* xrow = (const unsigned int*)(xh + (size_t)(row0 + lm) * KD);

  // A 16x32 bf16 layout per step: lanes0-15 K=0..7 & 16..23 ; lanes16-31 K=8..15 & 24..31
  BFrag a[4];
#pragma unroll
  for (int kk = 0; kk < 4; ++kk) {
    const int kb = kk << 5;
    uint4 a0 = *(const uint4*)(xrow + ((kb + hh * 8) >> 1));
    uint4 a1 = *(const uint4*)(xrow + ((kb + 16 + hh * 8) >> 1));
    a[kk].u[0] = a0.x; a[kk].u[1] = a0.y; a[kk].u[2] = a0.z; a[kk].u[3] = a0.w;
    a[kk].u[4] = a1.x; a[kk].u[5] = a1.y; a[kk].u[6] = a1.z; a[kk].u[7] = a1.w;
  }

  for (int ct = 0; ct < (Nout >> 4); ++ct) {
    const int col0 = ct << 4;
    const unsigned int* wcol = (const unsigned int*)(wcm + (size_t)(col0 + lm) * KD);
    v8f c = {0.f, 0.f, 0.f, 0.f, 0.f, 0.f, 0.f, 0.f};
#pragma unroll
    for (int kk = 0; kk < 4; ++kk) {
      const int kb = kk << 5;
      // B 32x16 bf16 layout: lanes0-15 K=0..7 & 16..23 ; lanes16-31 K=8..15 & 24..31 (2 per VGPR)
      BFrag b;
      uint4 b0 = *(const uint4*)(wcol + ((kb + hh * 8) >> 1));
      uint4 b1 = *(const uint4*)(wcol + ((kb + 16 + hh * 8) >> 1));
      b.u[0] = b0.x; b.u[1] = b0.y; b.u[2] = b0.z; b.u[3] = b0.w;
      b.u[4] = b1.x; b.u[5] = b1.y; b.u[6] = b1.z; b.u[7] = b1.w;
      c = __builtin_amdgcn_wmma_f32_16x16x32_bf16(false, a[kk].v, false, b.v,
                                                  (short)0, c, false, false);
    }
    // C/D layout: n = lane&15, m = (lane>>4)*8 + v
#pragma unroll
    for (int v = 0; v < 8; ++v) {
      size_t idx = (size_t)(row0 + hh * 8 + v) * Nout + col0 + lm;
      agg[idx] = c[v];
      ybf[idx] = f2bf(c[v]);
    }
  }
}

// ---------------- edge scatter: agg[col] += y[row]  (bf16 gather, f32 atomics) ----
// Each wave loads 32 edge indices coalesced, then per edge broadcasts the
// row/col via shfl; the 32 lanes cover one full feature row -> one 256B L2
// read + one coalesced burst of global_atomic_add_f32 per edge.
template <int D>
__global__ void __launch_bounds__(256) k_scatter(const int* __restrict__ rows,
                                                 const int* __restrict__ cols,
                                                 const unsigned short* __restrict__ ybf,
                                                 float* __restrict__ agg) {
  constexpr int CH = D / 32;                 // floats per lane: 4 (D=128) or 2 (D=64)
  const int lane  = threadIdx.x & 31;
  const int ebase = (blockIdx.x * (256 >> 5) + (threadIdx.x >> 5)) << 5;
  const int r = rows[ebase + lane];
  const int c = cols[ebase + lane];
#pragma unroll
  for (int i = 0; i < 32; ++i) {
    const int ri = __shfl(r, i);
    const int ci = __shfl(c, i);
    const unsigned short* src = ybf + (size_t)ri * D + lane * CH;
    float* dst = agg + (size_t)ci * D + lane * CH;
    if constexpr (CH == 4) {
      union { unsigned long long q; unsigned short s[4]; } pk;
      pk.q = *(const unsigned long long*)src;
      unsafeAtomicAdd(dst + 0, bf2f(pk.s[0]));
      unsafeAtomicAdd(dst + 1, bf2f(pk.s[1]));
      unsafeAtomicAdd(dst + 2, bf2f(pk.s[2]));
      unsafeAtomicAdd(dst + 3, bf2f(pk.s[3]));
    } else {
      union { unsigned int q; unsigned short s[2]; } pk;
      pk.q = *(const unsigned int*)src;
      unsafeAtomicAdd(dst + 0, bf2f(pk.s[0]));
      unsafeAtomicAdd(dst + 1, bf2f(pk.s[1]));
    }
  }
}

// ---------------- finalize: h = relu(agg*dinv + b); emit next bf16 input = h*dinv --
__global__ void __launch_bounds__(256) k_fin_relu(const float* __restrict__ agg,
                                                  const float* __restrict__ dinv,
                                                  const float* __restrict__ bias,
                                                  unsigned short* __restrict__ xh) {
  int t = blockIdx.x * 256 + threadIdx.x;    // NN*KD
  float di = dinv[t >> 7];
  float v  = fmaf(agg[t], di, bias[t & 127]);
  v = v > 0.f ? v : 0.f;
  xh[t] = f2bf(v * di);                      // pre-scale for next layer's GEMM
}
__global__ void __launch_bounds__(256) k_fin_out(const float* __restrict__ agg,
                                                 const float* __restrict__ dinv,
                                                 const float* __restrict__ bias,
                                                 float* __restrict__ out) {
  int t = blockIdx.x * 256 + threadIdx.x;    // NN*64
  out[t] = fmaf(agg[t], dinv[t >> 6], bias[t & 63]);
}

extern "C" void kernel_launch(void* const* d_in, const int* in_sizes, int n_in,
                              void* d_out, int out_size, void* d_ws, size_t ws_size,
                              hipStream_t stream) {
  (void)in_sizes; (void)n_in; (void)out_size; (void)ws_size;
  const float* x  = (const float*)d_in[0];
  const int*   ei = (const int*)d_in[1];    // [2, NE]: rows = ei, cols = ei+NE
  const float* W1 = (const float*)d_in[2];
  const float* b1 = (const float*)d_in[3];
  const float* W2 = (const float*)d_in[4];
  const float* b2 = (const float*)d_in[5];
  const float* W3 = (const float*)d_in[6];
  const float* b3 = (const float*)d_in[7];

  char* p = (char*)d_ws;
  auto alloc = [&](size_t bytes) { char* q = p; p += (bytes + 255) & ~(size_t)255; return q; };
  float*          dinv = (float*)alloc((size_t)NN * 4);            // deg -> dinv in place
  unsigned short* xh   = (unsigned short*)alloc((size_t)NN * KD * 2);
  unsigned short* ybf  = (unsigned short*)alloc((size_t)NN * KD * 2);
  float*          agg  = (float*)alloc((size_t)NN * KD * 4);
  unsigned short* wcm  = (unsigned short*)alloc((size_t)KD * KD * 2);

  const int T = 256;
  // degrees -> dinv (shared by all 3 layers)
  k_deg_init <<<(NN + T - 1) / T, T, 0, stream>>>(dinv);
  k_deg_count<<<(NE + T - 1) / T, T, 0, stream>>>(ei + NE, dinv);
  k_dinv     <<<(NN + T - 1) / T, T, 0, stream>>>(dinv);

  // layer 1
  k_cvt_w       <<<(KD * KD) / T, T, 0, stream>>>(W1, wcm, KD);
  k_cvt_x       <<<(NN * KD) / T, T, 0, stream>>>(x, dinv, xh);
  k_gemm_wmma   <<<NN / 16, 32, 0, stream>>>(xh, wcm, ybf, agg, KD);
  k_scatter<128><<<NE / T, T, 0, stream>>>(ei, ei + NE, ybf, agg);
  k_fin_relu    <<<(NN * KD) / T, T, 0, stream>>>(agg, dinv, b1, xh);

  // layer 2
  k_cvt_w       <<<(KD * KD) / T, T, 0, stream>>>(W2, wcm, KD);
  k_gemm_wmma   <<<NN / 16, 32, 0, stream>>>(xh, wcm, ybf, agg, KD);
  k_scatter<128><<<NE / T, T, 0, stream>>>(ei, ei + NE, ybf, agg);
  k_fin_relu    <<<(NN * KD) / T, T, 0, stream>>>(agg, dinv, b2, xh);

  // layer 3 (Dout = 64)
  k_cvt_w       <<<(64 * KD) / T, T, 0, stream>>>(W3, wcm, 64);
  k_gemm_wmma   <<<NN / 16, 32, 0, stream>>>(xh, wcm, ybf, agg, 64);
  k_scatter<64> <<<NE / T, T, 0, stream>>>(ei, ei + NE, ybf, agg);
  k_fin_out     <<<(NN * 64) / T, T, 0, stream>>>(agg, dinv, b3, (float*)d_out);
}
